// LongformerClassifier_20942260535786
// MI455X (gfx1250) — matrix operand order repaired and checked by reading, not compiled, not run on hardware
//
#include <hip/hip_runtime.h>
#include <hip/hip_bf16.h>
#include <math.h>
#include <stdint.h>

// ---------------------------------------------------------------- constants
static constexpr int kS   = 2048;   // sequence
static constexpr int kH   = 768;    // hidden
static constexpr int kL   = 12;     // layers
static constexpr int kHN  = 12;     // heads
static constexpr int kD   = 64;     // head dim
static constexpr int kFF  = 3072;   // ffn dim
static constexpr int kY   = 50;     // classes
static constexpr int kC   = 256;    // chunk
static constexpr int kW1S = 256;    // band half width

typedef __bf16 bf16_t;
typedef __attribute__((ext_vector_type(16))) __bf16 v16bf;
typedef __attribute__((ext_vector_type(8)))  float  v8f;
typedef __attribute__((ext_vector_type(4)))  unsigned int v4u;
typedef __attribute__((ext_vector_type(8)))  int v8i;
typedef __attribute__((ext_vector_type(4)))  int v4i;

#ifndef __has_builtin
#define __has_builtin(x) 0
#endif
#if __has_builtin(__builtin_amdgcn_tensor_load_to_lds) && \
    __has_builtin(__builtin_amdgcn_s_wait_tensorcnt)
#define HAVE_TDM 1
#if defined(__clang_major__) && (__clang_major__ >= 23)
#pragma message("CDNA5 probe: TDM tensor_load_to_lds ENABLED (6-arg clang>=23 form)")
#else
#pragma message("CDNA5 probe: TDM tensor_load_to_lds ENABLED (5-arg ROCm/clang-22 form)")
#endif
#else
#define HAVE_TDM 0
#if defined(__AMDGCN__)
#pragma message("CDNA5 probe: TDM builtin NOT found on device pass; manual LDS staging")
#endif
#endif

union Frag { v16bf v; unsigned int u[8]; };

__device__ inline v8f wmma_bf16(const Frag& a, const Frag& b, v8f c) {
  // D = A(16x32 bf16) * B(32x16 bf16) + C(16x16 f32)
  return __builtin_amdgcn_wmma_f32_16x16x32_bf16(
      /*neg_a=*/false, a.v, /*neg_b=*/false, b.v,
      /*c_mod=*/(short)0, c, /*reuse_a=*/false, /*reuse_b=*/false);
}

// A-matrix (16x32, 16-bit) per-lane dword j -> starting K (pairs K,K+1).
__device__ inline int kbaseA(int j, int khalf) {
  return (j < 4) ? (khalf * 8 + 2 * j) : (16 + khalf * 8 + 2 * (j - 4));
}
// B-matrix (32x16, 16-bit): same K order as A (dword j -> starting K).
__device__ inline int kbaseB(int j, int khalf) { return kbaseA(j, khalf); }

__device__ inline unsigned int ld32(const bf16_t* p) {
  return *reinterpret_cast<const unsigned int*>(p);
}

__device__ inline float rowmax16(float v) {
  for (int o = 1; o < 16; o <<= 1) v = fmaxf(v, __shfl_xor(v, o, 32));
  return v;
}
__device__ inline float rowsum16(float v) {
  for (int o = 1; o < 16; o <<= 1) v += __shfl_xor(v, o, 32);
  return v;
}

__device__ inline float block_reduce_sum(float v, float* red) {
  const int tid = threadIdx.x;
  red[tid] = v; __syncthreads();
  for (int s = 128; s > 0; s >>= 1) {
    if (tid < s && tid + s < 256) red[tid] += red[tid + s];
    __syncthreads();
  }
  float r = red[0]; __syncthreads();
  return r;
}

#if HAVE_TDM
// Issue (no wait) a TDM DMA of a (rows x 32) bf16 tile of an (M x K) bf16
// tensor into LDS. D# per CDNA5 ISA (2D tensor).
__device__ inline void tdm_issue_tile(const bf16_t* gtile, unsigned int lds_off,
                                      int K, int M, int rows) {
  unsigned long long ga = (unsigned long long)(uintptr_t)gtile;
  v4u g0;
  g0[0] = 1u;                                   // count=1, user descriptor
  g0[1] = lds_off;                              // lds_addr (bytes)
  g0[2] = (unsigned int)(ga & 0xffffffffull);   // global_addr[31:0]
  g0[3] = (unsigned int)((ga >> 32) & 0x01ffffffull) | (2u << 30);  // type=2
  v8i g1;
  g1[0] = (int)(1u << 16);                      // data_size=1 -> 2 bytes/elem
  g1[1] = (int)(((unsigned int)K & 0xffffu) << 16);          // tensor_dim0 lo16
  g1[2] = (int)(((unsigned int)K >> 16) |
                (((unsigned int)M & 0xffffu) << 16));        // dim0 hi | dim1 lo
  g1[3] = (int)(((unsigned int)M >> 16) | (32u << 16));      // dim1 hi | tile_dim0=32
  g1[4] = (int)(unsigned int)rows;              // tile_dim1=rows, tile_dim2=0
  g1[5] = (int)K;                               // tensor_dim0_stride lo32
  g1[6] = 0;                                    // stride hi16 | dim1_stride lo16
  g1[7] = 0;
  v4i z4 = (v4i)0;
#if defined(__clang_major__) && (__clang_major__ >= 23)
  v8i z8 = (v8i)0;
  __builtin_amdgcn_tensor_load_to_lds(g0, g1, z4, z4, z8, 0);
#else
  __builtin_amdgcn_tensor_load_to_lds(g0, g1, z4, z4, 0);
#endif
}
#endif

// ---------------------------------------------------------------- embedding
__global__ __launch_bounds__(256)
void k_embed(const int* __restrict__ ids, const int* __restrict__ tt,
             const float* __restrict__ wemb, const float* __restrict__ pemb,
             const float* __restrict__ temb, const float* __restrict__ g,
             const float* __restrict__ b, float* __restrict__ hf,
             bf16_t* __restrict__ hb) {
  __shared__ float red[256];
  const int s = blockIdx.x;
  const int tid = threadIdx.x;
  const float* wr = wemb + (size_t)ids[s] * kH;
  const float* pr = pemb + (size_t)s * kH;
  const float* tr = temb + (size_t)tt[s] * kH;
  float x[3];
  for (int i = 0; i < 3; ++i) {
    int idx = tid + i * 256;
    x[i] = wr[idx] + pr[idx] + tr[idx];
  }
  float mu = block_reduce_sum(x[0] + x[1] + x[2], red) * (1.0f / kH);
  float vs = 0.f;
  for (int i = 0; i < 3; ++i) { float d = x[i] - mu; vs += d * d; }
  float var = block_reduce_sum(vs, red) * (1.0f / kH);
  float rstd = rsqrtf(var + 1e-5f);
  for (int i = 0; i < 3; ++i) {
    int idx = tid + i * 256;
    float y = (x[i] - mu) * rstd * g[idx] + b[idx];
    hf[(size_t)s * kH + idx] = y;
    hb[(size_t)s * kH + idx] = (bf16_t)y;
  }
}

// ------------------------------------------------------- residual + LayerNorm
__global__ __launch_bounds__(256)
void k_addln(const float* resid, const float* delta,
             const float* __restrict__ g, const float* __restrict__ b,
             float* outf, bf16_t* __restrict__ outb) {
  __shared__ float red[256];
  const int s = blockIdx.x;
  const int tid = threadIdx.x;
  float x[3];
  for (int i = 0; i < 3; ++i) {
    int idx = tid + i * 256;
    x[i] = resid[(size_t)s * kH + idx] + delta[(size_t)s * kH + idx];
  }
  float mu = block_reduce_sum(x[0] + x[1] + x[2], red) * (1.0f / kH);
  float vs = 0.f;
  for (int i = 0; i < 3; ++i) { float d = x[i] - mu; vs += d * d; }
  float var = block_reduce_sum(vs, red) * (1.0f / kH);
  float rstd = rsqrtf(var + 1e-5f);
  for (int i = 0; i < 3; ++i) {
    int idx = tid + i * 256;
    float y = (x[i] - mu) * rstd * g[idx] + b[idx];
    outf[(size_t)s * kH + idx] = y;
    outb[(size_t)s * kH + idx] = (bf16_t)y;
  }
}

// ------------------------------------------------ WMMA GEMM  C = A*W + bias
// A: MxK bf16 row-major, staged into double-buffered LDS by the Tensor Data
// Mover (pipelined: DMA for tile i+1 overlaps WMMA on tile i). W: KxN fp32
// row-major, converted fp32->bf16 during LDS transpose staging.
// Block: 256 threads (8 waves), tile 256x64, BLK_K = 32; wave = 32x64 -> 8 wmma.
template <bool GELU>
__global__ __launch_bounds__(256)
void k_gemm(const bf16_t* __restrict__ A, const float* __restrict__ W,
            const float* __restrict__ bias, float* __restrict__ Cf,
            bf16_t* __restrict__ Cb, int M, int N, int K) {
  __shared__ __align__(16) bf16_t sA[2][256][32];   // 32 KB double buffered
  __shared__ __align__(16) bf16_t sB[2][64][32];    // 8 KB, transposed [n][k]
  const int tid = threadIdx.x;
  const int wave = tid >> 5, lane = tid & 31;
  const int l16 = lane & 15, khalf = lane >> 4;
  const int m0 = blockIdx.x * 256, n0 = blockIdx.y * 64;
  const int nsteps = K / 32;

  v8f acc[2][4];
  for (int mt = 0; mt < 2; ++mt)
    for (int nt = 0; nt < 4; ++nt)
      for (int i = 0; i < 8; ++i) acc[mt][nt][i] = 0.f;

#if HAVE_TDM
  if (wave == 0)
    tdm_issue_tile(A + (size_t)m0 * K,
                   (unsigned int)(uintptr_t)(void*)&sA[0][0][0], K, M, 256);
#endif

  for (int step = 0; step < nsteps; ++step) {
    const int k0 = step * 32;
    const int buf = step & 1;
    // stage W (32x64 fp32 -> bf16, transposed into sB[buf][n][k])
    {
      int kr = tid >> 3, ng = (tid & 7) * 8;
      const float4* wp =
          reinterpret_cast<const float4*>(W + (size_t)(k0 + kr) * N + n0 + ng);
      float4 a4 = wp[0], b4 = wp[1];
      float vals[8] = {a4.x, a4.y, a4.z, a4.w, b4.x, b4.y, b4.z, b4.w};
      for (int j = 0; j < 8; ++j) sB[buf][ng + j][kr] = (bf16_t)vals[j];
      if (k0 + 32 < K)
        __builtin_prefetch(W + (size_t)(k0 + 32 + kr) * N + n0 + ng, 0, 1);
    }
#if !HAVE_TDM
    for (int gidx = tid; gidx < 1024; gidx += 256) {
      int r = gidx >> 2, cg = (gidx & 3) * 8;
      uint4 d = *reinterpret_cast<const uint4*>(A + (size_t)(m0 + r) * K + k0 + cg);
      *reinterpret_cast<uint4*>(&sA[buf][r][cg]) = d;
    }
#endif
    __syncthreads();  // all waves done with compute step-1; staging published
#if HAVE_TDM
    if (wave == 0) {
      if (step + 1 < nsteps) {
        // Kick DMA for the next tile into the alternate buffer, then wait
        // only for the *older* DMA (this step's tile): pipelined overlap.
        tdm_issue_tile(A + (size_t)m0 * K + (k0 + 32),
                       (unsigned int)(uintptr_t)(void*)&sA[buf ^ 1][0][0],
                       K, M, 256);
        __builtin_amdgcn_s_wait_tensorcnt(1);
      } else {
        __builtin_amdgcn_s_wait_tensorcnt(0);
      }
    }
    __syncthreads();  // publish TDM completion of sA[buf] to all waves
#endif

    Frag fb[4];
    for (int nt = 0; nt < 4; ++nt) {
      const int col = nt * 16 + l16;
      for (int j = 0; j < 8; ++j)
        fb[nt].u[j] = ld32(&sB[buf][col][kbaseB(j, khalf)]);
    }
    for (int mt = 0; mt < 2; ++mt) {
      Frag fa;
      const int arow = wave * 32 + mt * 16 + l16;
      for (int j = 0; j < 8; ++j)
        fa.u[j] = ld32(&sA[buf][arow][kbaseA(j, khalf)]);
      for (int nt = 0; nt < 4; ++nt)
        acc[mt][nt] = wmma_bf16(fa, fb[nt], acc[mt][nt]);
    }
    // No trailing barrier: next iteration stages into the other buffer, and
    // its barrier precedes any overwrite of the buffers read here.
  }

  for (int mt = 0; mt < 2; ++mt)
    for (int nt = 0; nt < 4; ++nt)
      for (int r = 0; r < 8; ++r) {
        int row = m0 + wave * 32 + mt * 16 + 8 * khalf + r;
        int col = n0 + nt * 16 + l16;
        float vv = acc[mt][nt][r] + bias[col];
        if (GELU) vv = 0.5f * vv * (1.0f + erff(vv * 0.70710678118f));
        if (Cf) Cf[(size_t)row * N + col] = vv;
        if (Cb) Cb[(size_t)row * N + col] = (bf16_t)vv;
      }
}

// --------------------------------------------- banded flash attention (WMMA)
// grid: 12 heads * 8 chunks * 2 halves = 192 blocks; wave = 16 query rows.
__global__ __launch_bounds__(256)
void k_attn(const bf16_t* __restrict__ q, const bf16_t* __restrict__ k,
            const bf16_t* __restrict__ v, const int* __restrict__ am,
            bf16_t* __restrict__ out) {
  __shared__ __align__(16) bf16_t sVT[64][32];     // 4 KB (shared across waves)
  __shared__ __align__(16) bf16_t sP[8][16][32];   // 8 KB (per-wave P tile)
  const int tid = threadIdx.x, wave = tid >> 5, lane = tid & 31;
  const int l16 = lane & 15, khalf = lane >> 4;
  const int bid = blockIdx.x;
  const int head = bid >> 4;
  const int rem = bid & 15;
  const int chunk = rem >> 1;
  const int rg = (rem & 1) * 8 + wave;  // rowgroup 0..15

  // Q fragments (K = head dim, split 0..31 / 32..63)
  Frag aq0, aq1;
  {
    const int qrow = chunk * kC + rg * 16 + l16;
    const bf16_t* qp = q + (size_t)qrow * kH + head * kD;
    for (int j = 0; j < 8; ++j) {
      int kb = kbaseA(j, khalf);
      aq0.u[j] = ld32(qp + kb);
      aq1.u[j] = ld32(qp + 32 + kb);
    }
  }

  v8f acc[4];
  for (int t = 0; t < 4; ++t)
    for (int i = 0; i < 8; ++i) acc[t][i] = 0.f;
  float mrow[8], lrow[8];
  for (int r = 0; r < 8; ++r) { mrow[r] = -1e30f; lrow[r] = 0.f; }

  const float scale = 0.125f;  // 1/sqrt(64)
  for (int kb32 = 0; kb32 < 24; ++kb32) {
    const int j0 = kb32 * 32;
    // stage V tile transposed: sVT[d][key], 32 keys x 64 dims
    {
      int key = tid & 31, d0 = (tid >> 5) * 8;
      int kpos = chunk * kC - kC + j0 + key;
      int kr = min(max(kpos, 0), kS - 1);
      uint4 raw = *reinterpret_cast<const uint4*>(
          v + (size_t)kr * kH + head * kD + d0);
      const bf16_t* pr8 = reinterpret_cast<const bf16_t*>(&raw);
      for (int j = 0; j < 8; ++j) sVT[d0 + j][key] = pr8[j];
    }
    __syncthreads();

    float p0[8], p1[8];
    for (int kt = 0; kt < 2; ++kt) {
      const int jcol = j0 + kt * 16 + l16;
      const int kpos = chunk * kC - kC + jcol;
      const int kr = min(max(kpos, 0), kS - 1);
      const bf16_t* kp = k + (size_t)kr * kH + head * kD;
      Frag bk0, bk1;
      for (int j = 0; j < 8; ++j) {
        int kb = kbaseB(j, khalf);
        bk0.u[j] = ld32(kp + kb);
        bk1.u[j] = ld32(kp + 32 + kb);
      }
      v8f z;
      for (int i = 0; i < 8; ++i) z[i] = 0.f;
      v8f sc = wmma_bf16(aq0, bk0, z);
      sc = wmma_bf16(aq1, bk1, sc);
      const bool kmv = (kpos >= 0) && (kpos < kS) && (am[kr] != 0);
      float* dst = kt ? p1 : p0;
      for (int r = 0; r < 8; ++r) {
        int pc = rg * 16 + 8 * khalf + r;          // query pos in chunk
        int rel = jcol - kC - pc;                  // key - query offset
        bool band = (rel >= -kW1S) && (rel <= kW1S);
        dst[r] = (kmv && band) ? sc[r] * scale : -1e9f;
      }
    }

    // online softmax per row (16 lanes per half-wave hold one row's columns)
    for (int r = 0; r < 8; ++r) {
      float t = rowmax16(fmaxf(p0[r], p1[r]));
      float mnew = fmaxf(mrow[r], t);
      float rescale = __expf(mrow[r] - mnew);
      float e0 = __expf(p0[r] - mnew);
      float e1 = __expf(p1[r] - mnew);
      lrow[r] = lrow[r] * rescale + rowsum16(e0 + e1);
      mrow[r] = mnew;
      for (int nt = 0; nt < 4; ++nt) acc[nt][r] *= rescale;
      sP[wave][8 * khalf + r][l16] = (bf16_t)e0;        // transpose via LDS
      sP[wave][8 * khalf + r][16 + l16] = (bf16_t)e1;
    }

    // P(16x32) @ V(32x64) accumulate
    Frag pa;
    for (int j = 0; j < 8; ++j) pa.u[j] = ld32(&sP[wave][l16][kbaseA(j, khalf)]);
    for (int nt = 0; nt < 4; ++nt) {
      Frag bv_;
      const int d = nt * 16 + l16;
      for (int j = 0; j < 8; ++j) bv_.u[j] = ld32(&sVT[d][kbaseB(j, khalf)]);
      acc[nt] = wmma_bf16(pa, bv_, acc[nt]);
    }
    __syncthreads();
  }

  for (int nt = 0; nt < 4; ++nt)
    for (int r = 0; r < 8; ++r) {
      int row = chunk * kC + rg * 16 + 8 * khalf + r;
      int col = head * kD + nt * 16 + l16;
      out[(size_t)row * kH + col] = (bf16_t)(acc[nt][r] / lrow[r]);
    }
}

// ------------------------------------------------------------- pooling head
__global__ __launch_bounds__(256)
void k_pool(const float* __restrict__ x, const float* __restrict__ Uw,
            const float* __restrict__ fw, const float* __restrict__ fb,
            float* __restrict__ yout, float* __restrict__ alpha,
            float* __restrict__ mout) {
  __shared__ float sdot[kS];  // 8 KB
  __shared__ float red[256];
  const int yy = blockIdx.x, tid = threadIdx.x;
  const float* u = Uw + (size_t)yy * kH;
  for (int s = tid; s < kS; s += 256) {
    const float* xr = x + (size_t)s * kH;
    float d = 0.f;
    for (int h = 0; h < kH; ++h) d += u[h] * xr[h];
    sdot[s] = d;
  }
  __syncthreads();
  float lm = -1e30f;
  for (int s = tid; s < kS; s += 256) lm = fmaxf(lm, sdot[s]);
  red[tid] = lm; __syncthreads();
  for (int t = 128; t > 0; t >>= 1) {
    if (tid < t) red[tid] = fmaxf(red[tid], red[tid + t]);
    __syncthreads();
  }
  float mx = red[0]; __syncthreads();
  float ls = 0.f;
  for (int s = tid; s < kS; s += 256) {
    float e = __expf(sdot[s] - mx);
    sdot[s] = e;
    ls += e;
  }
  float tot = block_reduce_sum(ls, red);
  float inv = 1.f / tot;
  for (int s = tid; s < kS; s += 256) {
    float a = sdot[s] * inv;
    sdot[s] = a;
    alpha[(size_t)yy * kS + s] = a;
  }
  __syncthreads();
  float part = 0.f;
  for (int h = tid; h < kH; h += 256) {
    float a = 0.f;
    for (int s = 0; s < kS; ++s) a += sdot[s] * x[(size_t)s * kH + h];
    mout[(size_t)yy * kH + h] = a;
    part += fw[(size_t)yy * kH + h] * a;
  }
  float dotv = block_reduce_sum(part, red);
  if (tid == 0) yout[yy] = dotv + fb[yy];
}

__global__ void k_loss(const float* __restrict__ y, const float* __restrict__ tgt,
                       float* __restrict__ lossout) {
  __shared__ float red[64];
  const int tid = threadIdx.x;
  float v = 0.f;
  if (tid < kY) {
    float yi = y[tid], t = tgt[tid];
    v = fmaxf(yi, 0.f) - yi * t + log1pf(__expf(-fabsf(yi)));
  }
  red[tid] = v; __syncthreads();
  for (int s = 32; s > 0; s >>= 1) {
    if (tid < s) red[tid] += red[tid + s];
    __syncthreads();
  }
  if (tid == 0) lossout[0] = red[0] / (float)kY;
}

// ---------------------------------------------------------------- launcher
extern "C" void kernel_launch(void* const* d_in, const int* in_sizes, int n_in,
                              void* d_out, int out_size, void* d_ws,
                              size_t ws_size, hipStream_t stream) {
  (void)in_sizes; (void)n_in; (void)out_size; (void)ws_size;
  const int* ids = (const int*)d_in[0];
  const int* tt = (const int*)d_in[1];
  const int* am = (const int*)d_in[2];
  const float* target = (const float*)d_in[3];
  const float* wemb = (const float*)d_in[4];
  const float* pemb = (const float*)d_in[5];
  const float* temb = (const float*)d_in[6];
  const float* elng = (const float*)d_in[7];
  const float* elnb = (const float*)d_in[8];
  const float* Wq = (const float*)d_in[9];  const float* bq = (const float*)d_in[10];
  const float* Wk = (const float*)d_in[11]; const float* bk = (const float*)d_in[12];
  const float* Wv = (const float*)d_in[13]; const float* bv = (const float*)d_in[14];
  const float* Wo = (const float*)d_in[15]; const float* bo = (const float*)d_in[16];
  const float* ln1g = (const float*)d_in[17]; const float* ln1b = (const float*)d_in[18];
  const float* W1 = (const float*)d_in[19]; const float* b1 = (const float*)d_in[20];
  const float* W2 = (const float*)d_in[21]; const float* b2 = (const float*)d_in[22];
  const float* ln2g = (const float*)d_in[23]; const float* ln2b = (const float*)d_in[24];
  const float* Uw = (const float*)d_in[25];
  const float* fw = (const float*)d_in[26];
  const float* fbv = (const float*)d_in[27];

  char* ws = (char*)d_ws;
  size_t off = 0;
  auto take = [&](size_t bytes) -> void* {
    void* p = ws + off;
    off = (off + bytes + 255) & ~(size_t)255;
    return p;
  };
  float*  hf   = (float*)take((size_t)kS * kH * sizeof(float));
  bf16_t* hb   = (bf16_t*)take((size_t)kS * kH * sizeof(bf16_t));
  bf16_t* qb   = (bf16_t*)take((size_t)kS * kH * sizeof(bf16_t));
  bf16_t* kbuf = (bf16_t*)take((size_t)kS * kH * sizeof(bf16_t));
  bf16_t* vbuf = (bf16_t*)take((size_t)kS * kH * sizeof(bf16_t));
  bf16_t* ab   = (bf16_t*)take((size_t)kS * kH * sizeof(bf16_t));
  bf16_t* fbuf = (bf16_t*)take((size_t)kS * kFF * sizeof(bf16_t));
  float*  tf   = (float*)take((size_t)kS * kH * sizeof(float));

  k_embed<<<kS, 256, 0, stream>>>(ids, tt, wemb, pemb, temb, elng, elnb, hf, hb);

  const dim3 g768(kS / 256, kH / 64);
  const dim3 gff(kS / 256, kFF / 64);
  for (int l = 0; l < kL; ++l) {
    const size_t oHH = (size_t)l * kH * kH;
    const size_t oH = (size_t)l * kH;
    const size_t oHF = (size_t)l * kH * kFF;
    const size_t oF = (size_t)l * kFF;
    k_gemm<false><<<g768, 256, 0, stream>>>(hb, Wq + oHH, bq + oH, nullptr, qb, kS, kH, kH);
    k_gemm<false><<<g768, 256, 0, stream>>>(hb, Wk + oHH, bk + oH, nullptr, kbuf, kS, kH, kH);
    k_gemm<false><<<g768, 256, 0, stream>>>(hb, Wv + oHH, bv + oH, nullptr, vbuf, kS, kH, kH);
    k_attn<<<kHN * 16, 256, 0, stream>>>(qb, kbuf, vbuf, am, ab);
    k_gemm<false><<<g768, 256, 0, stream>>>(ab, Wo + oHH, bo + oH, tf, nullptr, kS, kH, kH);
    k_addln<<<kS, 256, 0, stream>>>(hf, tf, ln1g + oH, ln1b + oH, hf, hb);
    k_gemm<true><<<gff, 256, 0, stream>>>(hb, W1 + oHF, b1 + oF, nullptr, fbuf, kS, kFF, kH);
    k_gemm<false><<<g768, 256, 0, stream>>>(fbuf, W2 + oHF, b2 + oH, tf, nullptr, kS, kH, kFF);
    k_addln<<<kS, 256, 0, stream>>>(hf, tf, ln2g + oH, ln2b + oH, hf, hb);
  }

  float* yout = (float*)d_out;          // 50
  float* lossp = yout + kY;             // 1
  float* alpha = lossp + 1;             // 50*2048
  float* mo = alpha + (size_t)kY * kS;  // 50*768
  k_pool<<<kY, 256, 0, stream>>>(hf, Uw, fw, fbv, yout, alpha, mo);
  k_loss<<<1, 64, 0, stream>>>(yout, target, lossp);
}
